// MPNNIrregular_13726715478161
// MI455X (gfx1250) — matrix operand order repaired and checked by reading, not compiled, not run on hardware
//
#include <hip/hip_runtime.h>

// MPNN over irregular grid, CDNA5 (gfx1250) WMMA implementation.
// Dominant cost: edge message MLP (BS*NX*K = 262144 rows x (264->128->128)),
// ~158 GFLOP over 6 layers; all tensors (~30MB) are L2-resident on MI455X
// (192MB L2), so we run fp16 WMMA (f32 accum) with weights staged in LDS.
// Swish uses v_rcp_f32 (not IEEE divide) to keep GEMM epilogues thin.

typedef _Float16 half_t;
typedef __attribute__((ext_vector_type(16))) _Float16 v16h;
typedef __attribute__((ext_vector_type(8)))  float    v8f;

#define BSZ   4
#define NX    4096
#define SDIM  2
#define NPAR  5
#define HD    128
#define NL    6
#define KNN   16
#define DTC   0.1f
#define EPSC  1e-5f
#define KP_MSG 288      // 264 (2H+1+SD+NPAR) padded to multiple of 32
#define KP_EMB 32       // 8 (1+SD+NPAR) padded to 32
#define NNODES (BSZ*NX)

// swish(x) = x * sigmoid(x); v_exp_f32 + v_rcp_f32 (single-instruction recip)
__device__ __forceinline__ float swishf(float x){
  return x * __builtin_amdgcn_rcpf(1.0f + __expf(-x));
}

// ---- WMMA fragment helpers (16x16x32 f16, contiguous k order shared by A and B) ----
// A 16x32 (MxK), row-major tile in LDS/regs:
//  lanes 0-15 : row M=lane, K[0..15]
//  lanes 16-31: row M=lane-16, K[16..31]
__device__ __forceinline__ v16h frag_a(const half_t* tile, int ldk, int koff){
  int lane = threadIdx.x & 31;
  const half_t* p = tile + (size_t)(lane & 15) * ldk + koff + ((lane >> 4) << 4);
  v16h a;
#pragma unroll
  for (int i = 0; i < 8; ++i){ a[i] = p[i]; a[8+i] = p[8+i]; }
  return a;
}
// B 32x16 (KxN) from a transposed weight Wt[N][Kpad] (col n of W contiguous):
//  lanes 0-15 : col N=lane, K[koff..koff+15]; lanes 16-31: K[koff+16..koff+31]
__device__ __forceinline__ v16h frag_b(const half_t* wt, int ldk, int n0, int koff){
  int lane = threadIdx.x & 31;
  const half_t* p = wt + (size_t)(n0 + (lane & 15)) * ldk + koff + ((lane >> 4) << 4);
  v16h b;
#pragma unroll
  for (int i = 0; i < 16; ++i) b[i] = p[i];
  return b;
}
__device__ __forceinline__ v8f wmma32(v16h a, v16h b, v8f c){
  return __builtin_amdgcn_wmma_f32_16x16x32_f16(false, a, false, b, (short)0, c, false, false);
}
// D epilogue: lane (n, half) -> rows m0..m0+7; fused bias+swish, f16 store to LDS
__device__ __forceinline__ void d_swish_to_lds(v8f d, float bn, half_t* dst, int ldk, int n){
  int lane = threadIdx.x & 31;
  int m0 = (lane >> 4) << 3;
#pragma unroll
  for (int m = 0; m < 8; ++m)
    dst[(size_t)(m0 + m) * ldk + n] = (half_t)swishf(d[m] + bn);
}

// ---- weight convert: f32 [IN][OUT] (opt. layered via blockIdx.z) -> f16 [OUT][KP] ----
__global__ void convert_w(const float* __restrict__ w, half_t* __restrict__ wt,
                          int IN, int OUT, int KP){
  const float* wl = w + (size_t)blockIdx.z * IN * OUT;
  half_t* wtl = wt + (size_t)blockIdx.z * OUT * KP;
  for (int t = blockIdx.x * blockDim.x + threadIdx.x; t < OUT * KP;
       t += gridDim.x * blockDim.x){
    int o = t / KP, i = t % KP;
    wtl[t] = (i < IN) ? (half_t)wl[(size_t)i * OUT + o] : (half_t)0.0f;
  }
}

// ---- grid min/max per batch ----
__global__ void kminmax(const float* __restrict__ g, float* __restrict__ gmin,
                        float* __restrict__ gmax){
  __shared__ float s0[256], s1[256], s2[256], s3[256];
  int b = blockIdx.x, tid = threadIdx.x;
  float mn0 = 3e38f, mx0 = -3e38f, mn1 = 3e38f, mx1 = -3e38f;
  for (int i = tid; i < NX; i += 256){
    float x = g[((size_t)b*NX + i)*SDIM + 0];
    float y = g[((size_t)b*NX + i)*SDIM + 1];
    mn0 = fminf(mn0, x); mx0 = fmaxf(mx0, x);
    mn1 = fminf(mn1, y); mx1 = fmaxf(mx1, y);
  }
  s0[tid]=mn0; s1[tid]=mx0; s2[tid]=mn1; s3[tid]=mx1;
  __syncthreads();
  for (int o = 128; o > 0; o >>= 1){
    if (tid < o){
      s0[tid]=fminf(s0[tid],s0[tid+o]); s1[tid]=fmaxf(s1[tid],s1[tid+o]);
      s2[tid]=fminf(s2[tid],s2[tid+o]); s3[tid]=fmaxf(s3[tid],s3[tid+o]);
    }
    __syncthreads();
  }
  if (tid == 0){
    gmin[b*SDIM+0]=s0[0]; gmax[b*SDIM+0]=s1[0];
    gmin[b*SDIM+1]=s2[0]; gmax[b*SDIM+1]=s3[0];
  }
}

__global__ void kpos(const float* __restrict__ g, const float* __restrict__ gmin,
                     const float* __restrict__ gmax, float* __restrict__ pos){
  for (int t = blockIdx.x*blockDim.x + threadIdx.x; t < NNODES*SDIM;
       t += gridDim.x*blockDim.x){
    int node = t / SDIM, d = t % SDIM, b = node / NX;
    float mn = gmin[b*SDIM+d], mx = gmax[b*SDIM+d];
    pos[t] = (g[t] - mn) / (mx - mn);
  }
}

// ---- KNN top-16 (per-thread register insertion sort, LDS-tiled positions) ----
__global__ void kknn(const float* __restrict__ pos, int* __restrict__ nbr){
  __shared__ float sx[128], sy[128];
  int tid = threadIdx.x;
  int gid = blockIdx.x * 128 + tid;       // global node (blocks never straddle batches)
  int b = gid / NX, i = gid % NX;
  float xi = pos[(size_t)gid*SDIM], yi = pos[(size_t)gid*SDIM + 1];
  float bd[KNN]; int bi[KNN];
#pragma unroll
  for (int t = 0; t < KNN; ++t){ bd[t] = 3.0e38f; bi[t] = 0; }
  for (int jt = 0; jt < NX/128; ++jt){
    __syncthreads();
    int j0 = jt * 128;
    sx[tid] = pos[((size_t)b*NX + j0 + tid)*SDIM + 0];
    sy[tid] = pos[((size_t)b*NX + j0 + tid)*SDIM + 1];
    __syncthreads();
    for (int jj = 0; jj < 128; ++jj){
      int j = j0 + jj;
      float dx = xi - sx[jj], dy = yi - sy[jj];
      float d2 = dx*dx + dy*dy;
      if (j != i && d2 < bd[KNN-1]){
        bd[KNN-1] = d2; bi[KNN-1] = j;
#pragma unroll
        for (int t = KNN-1; t > 0; --t){
          if (bd[t] < bd[t-1]){
            float td = bd[t]; bd[t] = bd[t-1]; bd[t-1] = td;
            int ti = bi[t]; bi[t] = bi[t-1]; bi[t-1] = ti;
          }
        }
      }
    }
  }
#pragma unroll
  for (int t = 0; t < KNN; ++t) nbr[(size_t)gid*KNN + t] = bi[t];
}

// ---- embedding MLP: [u,pos,params](8->pad32) -> 128 -> 128 ----
__global__ void kembed(const float* __restrict__ u, const float* __restrict__ pr,
                       const float* __restrict__ pos,
                       const half_t* __restrict__ w1, const float* __restrict__ b1,
                       const half_t* __restrict__ w2, const float* __restrict__ b2,
                       float* __restrict__ fF32, half_t* __restrict__ fF16){
  extern __shared__ __align__(16) unsigned char smem[];
  half_t* sw1 = (half_t*)smem;              // 128*32
  half_t* sw2 = sw1 + HD*KP_EMB;            // 128*128
  half_t* sa  = sw2 + HD*HD;                // 8 waves * 16*32
  half_t* sh  = sa + 8*16*KP_EMB;           // 8 waves * 16*128
  int tid = threadIdx.x;
  for (int t = tid; t < HD*KP_EMB; t += blockDim.x) sw1[t] = w1[t];
  for (int t = tid; t < HD*HD;     t += blockDim.x) sw2[t] = w2[t];
  __syncthreads();
  int w = tid >> 5, lane = tid & 31;
  half_t* aw = sa + (size_t)w*16*KP_EMB;
  half_t* hw = sh + (size_t)w*16*HD;
  const int ngroups = NNODES/16/8;
  for (int g = blockIdx.x; g < ngroups; g += gridDim.x){
    int n0 = (g*8 + w) * 16;
    for (int t = lane; t < 16*KP_EMB; t += 32){
      int e = t / KP_EMB, c = t % KP_EMB, node = n0 + e;
      float v = 0.f;
      if (c == 0)              v = u[node];
      else if (c < 1+SDIM)     v = pos[(size_t)node*SDIM + (c-1)];
      else if (c < 1+SDIM+NPAR) v = pr[(size_t)node*NPAR + (c-1-SDIM)];
      aw[t] = (half_t)v;
    }
    for (int nt = 0; nt < HD/16; ++nt){
      v8f acc = {};
      acc = wmma32(frag_a(aw, KP_EMB, 0), frag_b(sw1, KP_EMB, nt*16, 0), acc);
      int n = nt*16 + (lane & 15);
      d_swish_to_lds(acc, b1[n], hw, HD, n);
    }
    for (int nt = 0; nt < HD/16; ++nt){
      v8f acc = {};
#pragma unroll
      for (int kc = 0; kc < HD/32; ++kc)
        acc = wmma32(frag_a(hw, HD, kc*32), frag_b(sw2, HD, nt*16, kc*32), acc);
      int n = nt*16 + (lane & 15);
      float bn = b2[n];
      int m0 = (lane >> 4) << 3;
#pragma unroll
      for (int m = 0; m < 8; ++m){
        float s = swishf(acc[m] + bn);
        size_t o = (size_t)(n0 + m0 + m)*HD + n;
        fF32[o] = s; fF16[o] = (half_t)s;
      }
    }
  }
}

// ---- message MLP + mean-over-K: one wave = one node's 16 edges ----
__global__ void kmsg(const half_t* __restrict__ f16, const float* __restrict__ u,
                     const float* __restrict__ pos, const float* __restrict__ pr,
                     const int* __restrict__ nbr,
                     const half_t* __restrict__ w1, const float* __restrict__ b1,
                     const half_t* __restrict__ w2, const float* __restrict__ b2,
                     float* __restrict__ agg){
  extern __shared__ __align__(16) unsigned char smem[];
  half_t* sw1 = (half_t*)smem;              // 288*128 = 73.7KB
  half_t* sw2 = sw1 + KP_MSG*HD;            // 128*128 = 32KB
  half_t* sa  = sw2 + HD*HD;                // 8 waves * 16*288
  half_t* sh  = sa + 8*16*KP_MSG;           // 8 waves * 16*128
  int tid = threadIdx.x;
  for (int t = tid; t < KP_MSG*HD; t += blockDim.x) sw1[t] = w1[t];
  for (int t = tid; t < HD*HD;     t += blockDim.x) sw2[t] = w2[t];
  __syncthreads();
  int w = tid >> 5, lane = tid & 31;
  half_t* aw = sa + (size_t)w*16*KP_MSG;
  half_t* hw = sh + (size_t)w*16*HD;
  const int ngroups = NNODES/8;             // 8 nodes per block iteration
  for (int g = blockIdx.x; g < ngroups; g += gridDim.x){
    int gi = g*8 + w;                       // global node id
    int b = gi / NX;
    const int* nrow = nbr + (size_t)gi*KNN;
    const half_t* fi = f16 + (size_t)gi*HD;
    float ui = u[gi];
    float pix = pos[(size_t)gi*SDIM], piy = pos[(size_t)gi*SDIM + 1];
    // assemble A: rows = 16 edges, cols = [f_i | f_j | du | dpos | var | pad]
    for (int t = lane; t < 16*KP_MSG; t += 32){
      int e = t / KP_MSG, c = t % KP_MSG;
      half_t v = (half_t)0.f;
      if (c < HD)            v = fi[c];
      else if (c < 2*HD){    int j = nrow[e];
                             v = f16[((size_t)(b*NX + j))*HD + (c - HD)]; }
      else if (c == 2*HD){   int j = nrow[e];
                             v = (half_t)(ui - u[b*NX + j]); }
      else if (c == 2*HD+1){ int j = nrow[e];
                             v = (half_t)(pix - pos[((size_t)(b*NX + j))*SDIM]); }
      else if (c == 2*HD+2){ int j = nrow[e];
                             v = (half_t)(piy - pos[((size_t)(b*NX + j))*SDIM + 1]); }
      else if (c < 2*HD+3+NPAR)
                             v = (half_t)pr[(size_t)gi*NPAR + (c - (2*HD+3))];
      aw[t] = v;
    }
    // GEMM1: 16x288 @ 288x128, fused bias+swish -> h
    for (int nt = 0; nt < HD/16; ++nt){
      v8f acc = {};
#pragma unroll
      for (int kc = 0; kc < KP_MSG/32; ++kc)
        acc = wmma32(frag_a(aw, KP_MSG, kc*32), frag_b(sw1, KP_MSG, nt*16, kc*32), acc);
      int n = nt*16 + (lane & 15);
      d_swish_to_lds(acc, b1[n], hw, HD, n);
    }
    // GEMM2: 16x128 @ 128x128, fused bias+swish + mean over the 16 edges
    for (int nt = 0; nt < HD/16; ++nt){
      v8f acc = {};
#pragma unroll
      for (int kc = 0; kc < HD/32; ++kc)
        acc = wmma32(frag_a(hw, HD, kc*32), frag_b(sw2, HD, nt*16, kc*32), acc);
      int n = nt*16 + (lane & 15);
      float bn = b2[n];
      float s = 0.f;
#pragma unroll
      for (int m = 0; m < 8; ++m) s += swishf(acc[m] + bn);
      s += __shfl_down(s, 16, 32);          // combine row halves (m 0-7 + 8-15)
      if (lane < 16) agg[(size_t)gi*HD + n] = s * (1.0f/16.0f);
    }
  }
}

// ---- update MLP + residual: rows = nodes ----
__global__ void kupd(const half_t* __restrict__ f16, const float* __restrict__ fF32,
                     const float* __restrict__ agg, const float* __restrict__ pr,
                     const half_t* __restrict__ w1, const float* __restrict__ b1,
                     const half_t* __restrict__ w2, const float* __restrict__ b2,
                     float* __restrict__ tmp){
  extern __shared__ __align__(16) unsigned char smem[];
  half_t* sw1 = (half_t*)smem;
  half_t* sw2 = sw1 + KP_MSG*HD;
  half_t* sa  = sw2 + HD*HD;
  half_t* sh  = sa + 8*16*KP_MSG;
  int tid = threadIdx.x;
  for (int t = tid; t < KP_MSG*HD; t += blockDim.x) sw1[t] = w1[t];
  for (int t = tid; t < HD*HD;     t += blockDim.x) sw2[t] = w2[t];
  __syncthreads();
  int w = tid >> 5, lane = tid & 31;
  half_t* aw = sa + (size_t)w*16*KP_MSG;
  half_t* hw = sh + (size_t)w*16*HD;
  const int ngroups = NNODES/16/8;
  for (int g = blockIdx.x; g < ngroups; g += gridDim.x){
    int n0 = (g*8 + w) * 16;
    for (int t = lane; t < 16*KP_MSG; t += 32){
      int e = t / KP_MSG, c = t % KP_MSG, node = n0 + e;
      half_t v = (half_t)0.f;
      if (c < HD)              v = f16[(size_t)node*HD + c];
      else if (c < 2*HD)       v = (half_t)agg[(size_t)node*HD + (c - HD)];
      else if (c < 2*HD+NPAR)  v = (half_t)pr[(size_t)node*NPAR + (c - 2*HD)];
      aw[t] = v;
    }
    for (int nt = 0; nt < HD/16; ++nt){
      v8f acc = {};
#pragma unroll
      for (int kc = 0; kc < KP_MSG/32; ++kc)
        acc = wmma32(frag_a(aw, KP_MSG, kc*32), frag_b(sw1, KP_MSG, nt*16, kc*32), acc);
      int n = nt*16 + (lane & 15);
      d_swish_to_lds(acc, b1[n], hw, HD, n);
    }
    for (int nt = 0; nt < HD/16; ++nt){
      v8f acc = {};
#pragma unroll
      for (int kc = 0; kc < HD/32; ++kc)
        acc = wmma32(frag_a(hw, HD, kc*32), frag_b(sw2, HD, nt*16, kc*32), acc);
      int n = nt*16 + (lane & 15);
      float bn = b2[n];
      int m0 = (lane >> 4) << 3;
#pragma unroll
      for (int m = 0; m < 8; ++m){
        size_t o = (size_t)(n0 + m0 + m)*HD + n;
        tmp[o] = fF32[o] + swishf(acc[m] + bn);   // residual
      }
    }
  }
}

// ---- instance norm over NX, per (batch, channel) ----
__global__ void knorm(const float* __restrict__ tmp, float* __restrict__ fF32,
                      half_t* __restrict__ f16){
  int b = blockIdx.x / HD, c = blockIdx.x % HD;
  __shared__ float ss[256], sq[256];
  int tid = threadIdx.x;
  float s = 0.f, q = 0.f;
  for (int i = tid; i < NX; i += 256){
    float v = tmp[((size_t)b*NX + i)*HD + c];
    s += v; q += v*v;
  }
  ss[tid] = s; sq[tid] = q; __syncthreads();
  for (int o = 128; o > 0; o >>= 1){
    if (tid < o){ ss[tid] += ss[tid+o]; sq[tid] += sq[tid+o]; }
    __syncthreads();
  }
  float mean = ss[0] * (1.0f/NX);
  float var  = sq[0] * (1.0f/NX) - mean*mean;
  float inv  = rsqrtf(var + EPSC);
  for (int i = tid; i < NX; i += 256){
    size_t o = ((size_t)b*NX + i)*HD + c;
    float v = (tmp[o] - mean) * inv;
    fF32[o] = v; f16[o] = (half_t)v;
  }
}

// ---- output head: swish(f@ow1+ob1)@ow2+ob2, out = u + DT*diff ----
__global__ void kout(const half_t* __restrict__ f16, const float* __restrict__ u,
                     const half_t* __restrict__ w1, const float* __restrict__ b1,
                     const float* __restrict__ w2, const float* __restrict__ b2,
                     float* __restrict__ out){
  extern __shared__ __align__(16) unsigned char smem[];
  half_t* sw1 = (half_t*)smem;               // 64*128
  half_t* sa  = sw1 + 64*HD;                 // 8 waves * 16*128
  float*  sg  = (float*)(sa + 8*16*HD);      // 8 waves * 16*64
  int tid = threadIdx.x;
  for (int t = tid; t < 64*HD; t += blockDim.x) sw1[t] = w1[t];
  __syncthreads();
  int w = tid >> 5, lane = tid & 31;
  half_t* aw = sa + (size_t)w*16*HD;
  float*  gw = sg + (size_t)w*16*64;
  const int ngroups = NNODES/16/8;
  for (int g = blockIdx.x; g < ngroups; g += gridDim.x){
    int n0 = (g*8 + w) * 16;
    for (int t = lane; t < 16*HD; t += 32)
      aw[t] = f16[(size_t)(n0 + t/HD)*HD + (t % HD)];
    for (int nt = 0; nt < 4; ++nt){
      v8f acc = {};
#pragma unroll
      for (int kc = 0; kc < HD/32; ++kc)
        acc = wmma32(frag_a(aw, HD, kc*32), frag_b(sw1, HD, nt*16, kc*32), acc);
      int n = nt*16 + (lane & 15);
      float bn = b1[n];
      int m0 = (lane >> 4) << 3;
#pragma unroll
      for (int m = 0; m < 8; ++m) gw[(m0 + m)*64 + n] = swishf(acc[m] + bn);
    }
    if (lane < 16){
      int node = n0 + lane;
      float s = 0.f;
#pragma unroll 8
      for (int c2 = 0; c2 < 64; ++c2) s += gw[lane*64 + c2] * w2[c2];
      out[node] = u[node] + DTC * (s + b2[0]);
    }
  }
}

extern "C" void kernel_launch(void* const* d_in, const int* in_sizes, int n_in,
                              void* d_out, int out_size, void* d_ws, size_t ws_size,
                              hipStream_t stream){
  (void)in_sizes; (void)n_in; (void)out_size; (void)ws_size;
  const float* inputs = (const float*)d_in[0];
  const float* casep  = (const float*)d_in[1];
  // d_in[2] mask unused
  const float* grid   = (const float*)d_in[3];
  const float* emb_w1 = (const float*)d_in[4];
  const float* emb_b1 = (const float*)d_in[5];
  const float* emb_w2 = (const float*)d_in[6];
  const float* emb_b2 = (const float*)d_in[7];
  const float* mw1 = (const float*)d_in[8];
  const float* mb1 = (const float*)d_in[9];
  const float* mw2 = (const float*)d_in[10];
  const float* mb2 = (const float*)d_in[11];
  const float* uw1 = (const float*)d_in[12];
  const float* ub1 = (const float*)d_in[13];
  const float* uw2 = (const float*)d_in[14];
  const float* ub2 = (const float*)d_in[15];
  const float* ow1 = (const float*)d_in[16];
  const float* ob1 = (const float*)d_in[17];
  const float* ow2 = (const float*)d_in[18];
  const float* ob2 = (const float*)d_in[19];
  float* out = (float*)d_out;

  // workspace partition (~32 MB total)
  char* ws = (char*)d_ws;
  size_t off = 0;
  auto take = [&](size_t bytes) -> char* {
    char* p = ws + off; off = (off + bytes + 255) & ~(size_t)255; return p;
  };
  float*  posF = (float*) take((size_t)NNODES*SDIM*4);
  float*  gmin = (float*) take((size_t)BSZ*SDIM*4);
  float*  gmax = (float*) take((size_t)BSZ*SDIM*4);
  int*    nbr  = (int*)   take((size_t)NNODES*KNN*4);
  float*  fF32 = (float*) take((size_t)NNODES*HD*4);
  half_t* fF16 = (half_t*)take((size_t)NNODES*HD*2);
  float*  agg  = (float*) take((size_t)NNODES*HD*4);
  float*  tmp  = (float*) take((size_t)NNODES*HD*4);
  half_t* eW1t = (half_t*)take((size_t)HD*KP_EMB*2);
  half_t* eW2t = (half_t*)take((size_t)HD*HD*2);
  half_t* mW1t = (half_t*)take((size_t)NL*HD*KP_MSG*2);
  half_t* mW2t = (half_t*)take((size_t)NL*HD*HD*2);
  half_t* uW1t = (half_t*)take((size_t)NL*HD*KP_MSG*2);
  half_t* uW2t = (half_t*)take((size_t)NL*HD*HD*2);
  half_t* oW1t = (half_t*)take((size_t)64*HD*2);

  // weights -> f16 transposed/padded
  convert_w<<<dim3(16,1,1),  256, 0, stream>>>(emb_w1, eW1t, 1+SDIM+NPAR, HD, KP_EMB);
  convert_w<<<dim3(64,1,1),  256, 0, stream>>>(emb_w2, eW2t, HD, HD, HD);
  convert_w<<<dim3(64,1,NL), 256, 0, stream>>>(mw1, mW1t, 2*HD+1+SDIM+NPAR, HD, KP_MSG);
  convert_w<<<dim3(64,1,NL), 256, 0, stream>>>(mw2, mW2t, HD, HD, HD);
  convert_w<<<dim3(64,1,NL), 256, 0, stream>>>(uw1, uW1t, 2*HD+NPAR, HD, KP_MSG);
  convert_w<<<dim3(64,1,NL), 256, 0, stream>>>(uw2, uW2t, HD, HD, HD);
  convert_w<<<dim3(16,1,1),  256, 0, stream>>>(ow1, oW1t, HD, 64, HD);

  kminmax<<<BSZ, 256, 0, stream>>>(grid, gmin, gmax);
  kpos<<<64, 256, 0, stream>>>(grid, gmin, gmax, posF);
  kknn<<<NNODES/128, 128, 0, stream>>>(posF, nbr);

  size_t smemEmb = (size_t)(HD*KP_EMB + HD*HD + 8*16*KP_EMB + 8*16*HD)*2;   // 80KB
  kembed<<<64, 256, smemEmb, stream>>>(inputs, casep, posF,
                                       eW1t, emb_b1, eW2t, emb_b2, fF32, fF16);

  size_t smemMsg = (size_t)(KP_MSG*HD + HD*HD + 8*16*KP_MSG + 8*16*HD)*2;   // 208KB
  for (int l = 0; l < NL; ++l){
    kmsg<<<256, 256, smemMsg, stream>>>(fF16, inputs, posF, casep, nbr,
        mW1t + (size_t)l*HD*KP_MSG, mb1 + (size_t)l*HD,
        mW2t + (size_t)l*HD*HD,     mb2 + (size_t)l*HD, agg);
    kupd<<<128, 256, smemMsg, stream>>>(fF16, fF32, agg, casep,
        uW1t + (size_t)l*HD*KP_MSG, ub1 + (size_t)l*HD,
        uW2t + (size_t)l*HD*HD,     ub2 + (size_t)l*HD, tmp);
    knorm<<<BSZ*HD, 256, 0, stream>>>(tmp, fF32, fF16);
  }

  size_t smemOut = (size_t)(64*HD + 8*16*HD)*2 + (size_t)8*16*64*4;         // 80KB
  kout<<<64, 256, smemOut, stream>>>(fF16, inputs, oW1t, ob1, ow2, ob2, out);
}
